// GraphNN_30331059044702
// MI455X (gfx1250) — compile-verified, matrix operand order checked by reading, not run on hardware
//
#include <hip/hip_runtime.h>
#include <hip/hip_bf16.h>
#include <cstdint>

typedef __attribute__((ext_vector_type(16))) _Float16 v16h;
typedef __attribute__((ext_vector_type(8)))  _Float16 v8h;
typedef __attribute__((ext_vector_type(8)))  float    v8f;
typedef __attribute__((ext_vector_type(4)))  float    v4f;

#define N_NODES 50000
#define N_PAIRS 400000
#define NE4 (4 * N_PAIRS)   /* all directed edges               */
#define NE2 (2 * N_PAIRS)   /* candidates with src > dst        */
#define NPAD (1 << 20)      /* bitonic pad (>= NE2)             */

// dynamic-LDS layout for the MLP kernel (bytes)
#define OFF_W0   0                       /* fc0 weights f16 [128][256] = 64 KB */
#define OFF_W1   (64 * 1024)             /* fc1 weights f16 [64][128]  = 16 KB */
#define OFF_SCR  (80 * 1024)             /* per-wave h0(f16)/h1(f32), 4 KB x 8 */
#define OFF_EA0  (112 * 1024)            /* per-wave ea0[16] f32               */
#define LDS_BYTES (112 * 1024 + 8 * 16 * 4)

// sched_group_barrier masks
#define SGB_MFMA  0x008   /* MFMA/WMMA  */
#define SGB_DSRD  0x100   /* DS read    */

// ---------------------------------------------------------------------------
// CDNA5 async global->LDS staging (ASYNCcnt path)
// ---------------------------------------------------------------------------
__device__ __forceinline__ void async_copy16(unsigned lds_addr, const void* g) {
    asm volatile("global_load_async_to_lds_b128 %0, %1, off"
                 :: "v"(lds_addr), "v"(g) : "memory");
}
__device__ __forceinline__ void wait_asynccnt0() {
    asm volatile("s_wait_asynccnt 0x0" ::: "memory");
}

// ---------------------------------------------------------------------------
// small helpers
// ---------------------------------------------------------------------------
__global__ void zero_f32(float* p, int n) {
    int i = blockIdx.x * blockDim.x + threadIdx.x;
    if (i < n) p[i] = 0.0f;
}

// ---------------------------------------------------------------------------
// GraphConv scatter: agg[dst] += x[src] * (ea0*ea1)   (fp32 global atomics)
// ---------------------------------------------------------------------------
__global__ void gcn_scatter(const float* __restrict__ x,
                            const int* __restrict__ edges,
                            const float* __restrict__ ea,
                            float* __restrict__ agg, int din) {
    int e = blockIdx.x * blockDim.x + threadIdx.x;
    if (e >= NE4) return;
    int s = edges[e];
    int d = edges[NE4 + e];
    float w = ea[2 * e] * ea[2 * e + 1];
    const float* xs = x + (size_t)s * din;
    float* ad = agg + (size_t)d * din;
    for (int k = 0; k < din; ++k) atomicAdd(&ad[k], xs[k] * w);
}

// ---------------------------------------------------------------------------
// GraphConv update: xout = relu(Wrel@agg + brel + Wroot@x)
// ---------------------------------------------------------------------------
__global__ void gcn_update(const float* __restrict__ x,
                           const float* __restrict__ agg,
                           const float* __restrict__ Wrel,
                           const float* __restrict__ brel,
                           const float* __restrict__ Wroot,
                           float* __restrict__ xout, int din, int dout) {
    int tid = blockIdx.x * blockDim.x + threadIdx.x;
    if (tid >= N_NODES * dout) return;
    int node = tid / dout;
    int o = tid - node * dout;
    const float* ar = agg + (size_t)node * din;
    const float* xr = x + (size_t)node * din;
    const float* wr = Wrel + (size_t)o * din;
    const float* wo = Wroot + (size_t)o * din;
    float acc = brel[o];
    for (int k = 0; k < din; ++k) acc += wr[k] * ar[k] + wo[k] * xr[k];
    xout[(size_t)node * dout + o] = acc > 0.0f ? acc : 0.0f;
}

// ---------------------------------------------------------------------------
// Edge filter + 52-bit stable lexsort key (src<<36 | dst<<20 | idx)
// ---------------------------------------------------------------------------
__global__ void build_keys(const int* __restrict__ edges,
                           const unsigned char* __restrict__ det,
                           unsigned long long* __restrict__ keys) {
    int i = blockIdx.x * blockDim.x + threadIdx.x;
    if (i >= NPAD) return;
    unsigned long long key = ~0ull;
    if (i < NE2) {
        int s = edges[i];
        int d = edges[NE4 + i];
        if (det[s] && det[d] && s > d)
            key = ((unsigned long long)s << 36) |
                  ((unsigned long long)d << 20) |
                  (unsigned long long)i;
    }
    keys[i] = key;
}

__global__ void bitonic_step(unsigned long long* keys, int kk, int jj) {
    int i = blockIdx.x * blockDim.x + threadIdx.x;
    int ixj = i ^ jj;
    if (ixj > i) {
        unsigned long long a = keys[i], b = keys[ixj];
        bool up = ((i & kk) == 0);
        if ((a > b) == up) { keys[i] = b; keys[ixj] = a; }
    }
}

// ---------------------------------------------------------------------------
// Repack fc weights to f16 for WMMA B-operand (row-major [out][K], col 128 of
// fc0 lifted out as the rank-1 ea0 term).
// ---------------------------------------------------------------------------
__global__ void pack_weights(const float* __restrict__ fc0W,
                             const float* __restrict__ fc1W,
                             _Float16* __restrict__ pfc0,
                             float* __restrict__ wmid,
                             _Float16* __restrict__ pfc1) {
    const int T0 = 128 * 256, T1 = 64 * 128, TOT = T0 + T1 + 128;
    for (int tid = blockIdx.x * blockDim.x + threadIdx.x; tid < TOT;
         tid += gridDim.x * blockDim.x) {
        if (tid < T0) {
            int o = tid >> 8, c = tid & 255;
            int sc = c < 128 ? c : c + 1;           // skip ea0 column
            pfc0[tid] = (_Float16)fc0W[o * 257 + sc];
        } else if (tid < T0 + T1) {
            int u = tid - T0;
            pfc1[u] = (_Float16)fc1W[u];
        } else {
            int o = tid - T0 - T1;
            wmid[o] = fc0W[o * 257 + 128];
        }
    }
}

// ---------------------------------------------------------------------------
// Edge-MLP: 8 waves/block, one wave = 16 sorted edges (128 edges/block).
// Weights async-staged to LDS once per block.  sched_group_barrier pins a
// software-pipelined [DSx4]([WMMA][DSx2])x7[WMMA] issue pattern per K-step so
// B fragments stay >=1 step ahead of the MMA pipe.
// ---------------------------------------------------------------------------
__global__ void __launch_bounds__(256)
mlp_kernel(const float* __restrict__ x,          // final GCN features, stride 128
           const int* __restrict__ edges,
           const float* __restrict__ ea,
           const unsigned long long* __restrict__ keys,
           const _Float16* __restrict__ pfc0, const float* __restrict__ wmid,
           const float* __restrict__ fc0b,
           const _Float16* __restrict__ pfc1, const float* __restrict__ fc1b,
           const float* __restrict__ outW, const float* __restrict__ outb,
           float* __restrict__ scores, int nEdges) {
    extern __shared__ __align__(16) char smem[];
    _Float16* w0 = (_Float16*)(smem + OFF_W0);   // [128][256]
    _Float16* w1 = (_Float16*)(smem + OFF_W1);   // [64][128]

    const int tid = threadIdx.x;
    const int lane = tid & 31;
    const int wv = tid >> 5;                     // wave in block (0..7)
    const int tile = blockIdx.x * 8 + wv;        // 16 edges / wave
    const int m = lane & 15;                     // row (edge in tile)
    const int kh = lane >> 4;                    // K-half select

    // ---- async-stage fc weights into LDS (whole block cooperates) ----------
    {
        const char* g0 = (const char*)pfc0;
        const char* g1 = (const char*)pfc1;
        for (int ofs = tid * 16; ofs < 64 * 1024; ofs += 256 * 16)
            async_copy16((unsigned)(uintptr_t)(smem + OFF_W0 + ofs), g0 + ofs);
        for (int ofs = tid * 16; ofs < 16 * 1024; ofs += 256 * 16)
            async_copy16((unsigned)(uintptr_t)(smem + OFF_W1 + ofs), g1 + ofs);
    }

    const int t = tile * 16 + m;                 // sorted-edge index
    const int tt = t < nEdges ? t : 0;
    const unsigned long long key = keys[tt];
    const int j = (int)(key & 0xFFFFFull);       // original edge id
    const int es = edges[j];
    const int ed = edges[NE4 + j];
    const float* xs = x + (size_t)es * 128;
    const float* xd = x + (size_t)ed * 128;

    float* eaw = (float*)(smem + OFF_EA0) + wv * 16;
    if (kh == 0) eaw[m] = ea[2 * j];             // ea0 (rank-1 term)

    wait_asynccnt0();
    __syncthreads();

    // ---- fc0: 16x256 @ 256x128, bias + ea0*W[:,128] folded into C init -----
    const int n = m;                             // D/B column = lane&15
    v8f acc[8];
    #pragma unroll
    for (int nt = 0; nt < 8; ++nt) {
        float b = fc0b[nt * 16 + n];
        float wm = wmid[nt * 16 + n];
        #pragma unroll
        for (int r = 0; r < 8; ++r)
            acc[nt][r] = b + eaw[r + 8 * kh] * wm;   // D row = r + 8*kh
    }

    // A fragment: feat[m][kb*32 + kh*8 + i] (i<8, and +16); software-pipelined
    v4f praw[4];
    union AF { v16h v; _Float16 e[16]; } A;
    auto issueA = [&](int kb) {
        const float* row = (kb < 4) ? xs : xd;
        const int rb = (kb & 3) * 32 + kh * 8;
        praw[0] = *(const v4f*)(row + rb);
        praw[1] = *(const v4f*)(row + rb + 4);
        praw[2] = *(const v4f*)(row + rb + 16);
        praw[3] = *(const v4f*)(row + rb + 20);
    };
    auto cvtA = [&]() {
        #pragma unroll
        for (int i = 0; i < 4; ++i) {
            A.e[i]      = (_Float16)praw[0][i];
            A.e[4 + i]  = (_Float16)praw[1][i];
            A.e[8 + i]  = (_Float16)praw[2][i];
            A.e[12 + i] = (_Float16)praw[3][i];
        }
    };
    issueA(0);
    cvtA();
    #pragma unroll
    for (int kb = 0; kb < 8; ++kb) {
        if (kb < 7) issueA(kb + 1);              // prefetch next A (raw f32)
        const int base = kb * 32 + kh * 8;
        union BF { v16h v; v8h h[2]; } B[8];
        #pragma unroll
        for (int nt = 0; nt < 8; ++nt) {
            const _Float16* wr = w0 + (nt * 16 + n) * 256;
            B[nt].h[0] = *(const v8h*)(wr + base);
            B[nt].h[1] = *(const v8h*)(wr + base + 16);
        }
        #pragma unroll
        for (int nt = 0; nt < 8; ++nt)
            acc[nt] = __builtin_amdgcn_wmma_f32_16x16x32_f16(
                false, A.v, false, B[nt].v, (short)0, acc[nt], false, false);
        if (kb < 7) cvtA();                      // fills WMMA->VALU hazard slots
        // pin pipelined issue order: B loads run >=1 fragment ahead of WMMAs
        __builtin_amdgcn_sched_group_barrier(SGB_DSRD, 4, 0);
        #pragma unroll
        for (int g = 0; g < 7; ++g) {
            __builtin_amdgcn_sched_group_barrier(SGB_MFMA, 1, 0);
            __builtin_amdgcn_sched_group_barrier(SGB_DSRD, 2, 0);
        }
        __builtin_amdgcn_sched_group_barrier(SGB_MFMA, 1, 0);
    }

    // ---- relu, h0 -> LDS as f16 [16][128] ----------------------------------
    _Float16* h0 = (_Float16*)(smem + OFF_SCR + wv * 4096);
    #pragma unroll
    for (int nt = 0; nt < 8; ++nt)
        #pragma unroll
        for (int r = 0; r < 8; ++r) {
            float v = acc[nt][r];
            h0[(r + 8 * kh) * 128 + nt * 16 + n] = (_Float16)(v > 0.0f ? v : 0.0f);
        }
    __syncthreads();

    // ---- fc1: 16x128 @ 128x64 ----------------------------------------------
    v8f acc2[4];
    #pragma unroll
    for (int nt = 0; nt < 4; ++nt) {
        float b = fc1b[nt * 16 + n];
        #pragma unroll
        for (int r = 0; r < 8; ++r) acc2[nt][r] = b;
    }
    #pragma unroll
    for (int kb = 0; kb < 4; ++kb) {
        const int base = kb * 32 + kh * 8;
        union AF2 { v16h v; v8h h[2]; } A2;
        A2.h[0] = *(const v8h*)&h0[m * 128 + base];
        A2.h[1] = *(const v8h*)&h0[m * 128 + base + 16];
        union BF { v16h v; v8h h[2]; } B[4];
        #pragma unroll
        for (int nt = 0; nt < 4; ++nt) {
            const _Float16* wr = w1 + (nt * 16 + n) * 128;
            B[nt].h[0] = *(const v8h*)(wr + base);
            B[nt].h[1] = *(const v8h*)(wr + base + 16);
        }
        #pragma unroll
        for (int nt = 0; nt < 4; ++nt)
            acc2[nt] = __builtin_amdgcn_wmma_f32_16x16x32_f16(
                false, A2.v, false, B[nt].v, (short)0, acc2[nt], false, false);
        // A2 + first two B fragments ahead, then overlap loads with WMMAs
        __builtin_amdgcn_sched_group_barrier(SGB_DSRD, 6, 0);
        #pragma unroll
        for (int g = 0; g < 3; ++g) {
            __builtin_amdgcn_sched_group_barrier(SGB_MFMA, 1, 0);
            __builtin_amdgcn_sched_group_barrier(SGB_DSRD, 2, 0);
        }
        __builtin_amdgcn_sched_group_barrier(SGB_MFMA, 1, 0);
    }
    __syncthreads();

    // ---- relu, h1 -> LDS as f32 [16][64]; split dot + cross-half combine ---
    float* h1 = (float*)(smem + OFF_SCR + wv * 4096);
    #pragma unroll
    for (int nt = 0; nt < 4; ++nt)
        #pragma unroll
        for (int r = 0; r < 8; ++r) {
            float v = acc2[nt][r];
            h1[(r + 8 * kh) * 64 + nt * 16 + n] = v > 0.0f ? v : 0.0f;
        }
    __syncthreads();

    {
        float spart = 0.0f;
        const float* hrow = h1 + m * 64 + kh * 32;
        const float* wrow = outW + kh * 32;
        #pragma unroll
        for (int k2 = 0; k2 < 32; ++k2) spart += hrow[k2] * wrow[k2];
        spart += __shfl_xor(spart, 16, 32);      // combine the two K-halves
        if (kh == 0 && t < nEdges) scores[t] = spart + outb[0];
    }
}

// ---------------------------------------------------------------------------
// Pair argmin + sigmoid + class select + edge output (edges stored as float)
// ---------------------------------------------------------------------------
__global__ void finalize(const unsigned long long* __restrict__ keys,
                         const float* __restrict__ scores,
                         const int* __restrict__ edges,
                         const float* __restrict__ ea,
                         float* __restrict__ out, int Mh) {
    int k = blockIdx.x * blockDim.x + threadIdx.x;
    if (k >= Mh) return;
    float s0 = scores[2 * k], s1 = scores[2 * k + 1];
    int j0 = (int)(keys[2 * k] & 0xFFFFFull);
    int j1 = (int)(keys[2 * k + 1] & 0xFFFFFull);
    int mi = (s1 < s0) ? 1 : 0;                      // argmin, first-on-tie
    float sel = mi ? s1 : s0;
    int jm = mi ? j1 : j0;
    out[k] = (float)edges[j0];                       // src row
    out[Mh + k] = (float)edges[NE4 + j0];            // dst row
    out[2 * Mh + k] = 1.0f / (1.0f + __expf(-sel));  // sigmoid(score)
    out[3 * Mh + k] = ea[2 * jm + 1];                // selected class
}

// ---------------------------------------------------------------------------
extern "C" void kernel_launch(void* const* d_in, const int* in_sizes, int n_in,
                              void* d_out, int out_size, void* d_ws, size_t ws_size,
                              hipStream_t stream) {
    const float* x0 = (const float*)d_in[0];
    const int* edges = (const int*)d_in[1];
    const float* eattr = (const float*)d_in[2];
    const unsigned char* det = (const unsigned char*)d_in[3];
    const float* gcW[3][3] = {
        {(const float*)d_in[4], (const float*)d_in[5], (const float*)d_in[6]},
        {(const float*)d_in[7], (const float*)d_in[8], (const float*)d_in[9]},
        {(const float*)d_in[10], (const float*)d_in[11], (const float*)d_in[12]}};
    const float* fc0W = (const float*)d_in[13];
    const float* fc0b = (const float*)d_in[14];
    const float* fc1W = (const float*)d_in[15];
    const float* fc1b = (const float*)d_in[16];
    const float* outW = (const float*)d_in[17];
    const float* outb = (const float*)d_in[18];

    // workspace carve-up (256-B aligned chunks)
    char* p = (char*)d_ws;
    auto take = [&](size_t bytes) {
        char* r = p;
        p += (bytes + 255) & ~(size_t)255;
        return r;
    };
    float* bufA = (float*)take((size_t)N_NODES * 128 * 4);
    float* bufB = (float*)take((size_t)N_NODES * 128 * 4);
    float* agg = (float*)take((size_t)N_NODES * 128 * 4);
    unsigned long long* keys = (unsigned long long*)take((size_t)NPAD * 8);
    _Float16* pfc0 = (_Float16*)take((size_t)128 * 256 * 2);
    _Float16* pfc1 = (_Float16*)take((size_t)64 * 128 * 2);
    float* wmid = (float*)take(128 * 4);
    float* scores = (float*)take((size_t)NE2 * 4);

    const int Mh = out_size / 4;       // outputs: 2*Mh edges + Mh sel + Mh cls
    const int nEdges = 2 * Mh;

    // 1) weight repack (f32 -> f16 WMMA layout)
    pack_weights<<<64, 256, 0, stream>>>(fc0W, fc1W, pfc0, wmid, pfc1);

    // 2) filter + sort keys (exact np.lexsort semantics)
    build_keys<<<NPAD / 256, 256, 0, stream>>>(edges, det, keys);
    for (int k = 2; k <= NPAD; k <<= 1)
        for (int jj = k >> 1; jj > 0; jj >>= 1)
            bitonic_step<<<NPAD / 256, 256, 0, stream>>>(keys, k, jj);

    // 3) GCN layers: 5->32->64->128
    const int dins[3] = {5, 32, 64};
    const int douts[3] = {32, 64, 128};
    const float* xin = x0;
    float* xout = bufA;
    for (int l = 0; l < 3; ++l) {
        int din = dins[l], dout = douts[l];
        int nz = N_NODES * din;
        zero_f32<<<(nz + 255) / 256, 256, 0, stream>>>(agg, nz);
        gcn_scatter<<<(NE4 + 255) / 256, 256, 0, stream>>>(xin, edges, eattr, agg, din);
        int nu = N_NODES * dout;
        gcn_update<<<(nu + 255) / 256, 256, 0, stream>>>(
            xin, agg, gcW[l][0], gcW[l][1], gcW[l][2], xout, din, dout);
        xin = xout;
        xout = (xout == bufA) ? bufB : bufA;
    }
    const float* xfinal = xin;         // stride 128

    // 4) WMMA edge-MLP (128 edges / block = 8 waves x 16 edges),
    //    weights async-staged into 112.5 KB of dynamic LDS per block.
    if (nEdges > 0) {
        int nBlocks = (nEdges + 127) / 128;
        mlp_kernel<<<nBlocks, 256, LDS_BYTES, stream>>>(
            xfinal, edges, eattr, keys, pfc0, wmid, fc0b, pfc1, fc1b,
            outW, outb, scores, nEdges);
        finalize<<<(Mh + 255) / 256, 256, 0, stream>>>(keys, scores, edges,
                                                       eattr, (float*)d_out, Mh);
    }
}
